// GNN_11166914969680
// MI455X (gfx1250) — matrix operand order repaired and checked by reading, not compiled, not run on hardware
//
#include <hip/hip_runtime.h>

typedef __attribute__((ext_vector_type(16))) __bf16 v16bf;
typedef __attribute__((ext_vector_type(8)))  float  v8f;

#define NEG_SLOPE 0.01f

// ---------------------------------------------------------------------------
// A-fragment (16-bit 16x32, CDNA5 ISA 7.12.2):
//   lane L (row M = L%16, half = L/16):
//   element e -> K = kb + half*8 + (e<8 ? e : 16 + (e-8))
// Four aligned float4 loads cover the pattern, then packed f32->bf16 converts.
// ---------------------------------------------------------------------------
__device__ __forceinline__ v16bf load_a_frag(const float* __restrict__ rowp,
                                             int kb, int half) {
    const float4* ap = (const float4*)(rowp + kb + half * 8);
    float4 q0 = ap[0];   // K + 0..3
    float4 q1 = ap[1];   // K + 4..7
    float4 q2 = ap[4];   // K + 16..19
    float4 q3 = ap[5];   // K + 20..23
    v16bf a;
    a[0]  = (__bf16)q0.x; a[1]  = (__bf16)q0.y; a[2]  = (__bf16)q0.z; a[3]  = (__bf16)q0.w;
    a[4]  = (__bf16)q1.x; a[5]  = (__bf16)q1.y; a[6]  = (__bf16)q1.z; a[7]  = (__bf16)q1.w;
    a[8]  = (__bf16)q2.x; a[9]  = (__bf16)q2.y; a[10] = (__bf16)q2.z; a[11] = (__bf16)q2.w;
    a[12] = (__bf16)q3.x; a[13] = (__bf16)q3.y; a[14] = (__bf16)q3.z; a[15] = (__bf16)q3.w;
    return a;
}

// ---------------------------------------------------------------------------
// Cooperative fill of fragment-swizzled W in LDS.
// Layout: sW[ ((kt*NT + nt)*32 + lane)*16 + e ]  (32 contiguous bytes per lane)
// Element value: W[ (kt*32 + (lane>>4)*8 + (e&7) + (e>>3)*16) * ldn + nt*16 + (lane&15) ]
// (B-fragment 32x16: lane holds col N=lane%16; element e -> K = kb + half*8 +
//  (e<8 ? e : 16 + (e-8)), matching the A-fragment order.)
// ---------------------------------------------------------------------------
__device__ __forceinline__ void fill_bw_lds(__bf16* sW, const float* __restrict__ W,
                                            int KT, int NT, int ldn, int nthreads) {
    const int total = KT * NT * 32 * 16;
    for (int i = threadIdx.x; i < total; i += nthreads) {
        int e    = i & 15;
        int lane = (i >> 4) & 31;
        int frag = i >> 9;                 // kt*NT + nt
        int nt   = frag % NT;
        int kt   = frag / NT;
        int k    = kt * 32 + (lane >> 4) * 8 + (e & 7) + (e >> 3) * 16;
        int n    = nt * 16 + (lane & 15);
        sW[i] = (__bf16)W[k * ldn + n];
    }
}

// B-fragment: one 32-byte aligned LDS vector load per fragment.
__device__ __forceinline__ v16bf load_b_frag(const __bf16* sW, int NT, int kt,
                                             int nt, int lane) {
    return *(const v16bf*)&sW[(((kt * NT) + nt) * 32 + lane) * 16];
}

// ---------------------------------------------------------------------------
// init: deg = 1 (self loop), zero both atomic accumulators
// ---------------------------------------------------------------------------
__global__ void gcn_init_kernel(float* deg, float* h1, float* h2agg, int N) {
    long long idx = (long long)blockIdx.x * blockDim.x + threadIdx.x;
    if (idx < (long long)N)       deg[idx]   = 1.0f;
    if (idx < (long long)N * 64)  h1[idx]    = 0.0f;
    if (idx < (long long)N * 32)  h2agg[idx] = 0.0f;
}

__global__ void gcn_deg_kernel(const int* __restrict__ dst, float* deg, int E) {
    int e = blockIdx.x * blockDim.x + threadIdx.x;
    if (e < E) atomicAdd(&deg[dst[e]], 1.0f);
}

__global__ void gcn_rsqrt_kernel(float* dinv, int N) {
    int i = blockIdx.x * blockDim.x + threadIdx.x;
    if (i < N) dinv[i] = rsqrtf(dinv[i]);   // deg >= 1 always (self loop)
}

// ---------------------------------------------------------------------------
// GEMM1: hpre[N,64] = x[N,256] @ W1[256,64]   (bf16 WMMA, f32 accum)
// 256 threads = 8 waves; each wave owns one 16-row M-tile x all 4 N-tiles.
// ---------------------------------------------------------------------------
__global__ void gcn_gemm1_wmma(const float* __restrict__ x,
                               const float* __restrict__ W1,
                               float* __restrict__ hpre, int N) {
    __shared__ __align__(32) __bf16 sW[8 * 4 * 32 * 16];   // 32 KB, swizzled
    fill_bw_lds(sW, W1, /*KT=*/8, /*NT=*/4, /*ldn=*/64, 256);
    __syncthreads();

    const int waveId = threadIdx.x >> 5;
    const int lane   = threadIdx.x & 31;
    const int mTile  = blockIdx.x * 8 + waveId;
    const int mTiles = (N + 15) >> 4;
    if (mTile >= mTiles) return;                            // wave-uniform exit

    const int half  = lane >> 4;
    const int lrow  = lane & 15;
    const int grow0 = mTile * 16;
    int arow = grow0 + lrow; if (arow > N - 1) arow = N - 1;
    const float* rowp = x + (size_t)arow * 256;

    v8f z = {0.f,0.f,0.f,0.f,0.f,0.f,0.f,0.f};
    v8f acc[4] = {z, z, z, z};

#pragma unroll
    for (int kt = 0; kt < 8; ++kt) {
        if (kt < 7) __builtin_prefetch(rowp + kt * 32 + 32, 0, 0);
        v16bf a = load_a_frag(rowp, kt * 32, half);
#pragma unroll
        for (int nt = 0; nt < 4; ++nt) {
            v16bf b = load_b_frag(sW, 4, kt, nt, lane);
            acc[nt] = __builtin_amdgcn_wmma_f32_16x16x32_bf16(
                false, a, false, b, (short)0, acc[nt], false, false);
        }
    }
    // C/D layout: VGPR r holds row M = r + half*8, col N = lane%16
#pragma unroll
    for (int nt = 0; nt < 4; ++nt)
#pragma unroll
        for (int r = 0; r < 8; ++r) {
            int grow = grow0 + half * 8 + r;
            if (grow < N) hpre[(size_t)grow * 64 + nt * 16 + lrow] = acc[nt][r];
        }
}

// ---------------------------------------------------------------------------
// GEMM2: h2pre[N,32] = h1[N,64] @ W2[64,32]
// ---------------------------------------------------------------------------
__global__ void gcn_gemm2_wmma(const float* __restrict__ h1,
                               const float* __restrict__ W2,
                               float* __restrict__ h2pre, int N) {
    __shared__ __align__(32) __bf16 sW[2 * 2 * 32 * 16];    // 4 KB, swizzled
    fill_bw_lds(sW, W2, /*KT=*/2, /*NT=*/2, /*ldn=*/32, 256);
    __syncthreads();

    const int waveId = threadIdx.x >> 5;
    const int lane   = threadIdx.x & 31;
    const int mTile  = blockIdx.x * 8 + waveId;
    const int mTiles = (N + 15) >> 4;
    if (mTile >= mTiles) return;

    const int half  = lane >> 4;
    const int lrow  = lane & 15;
    const int grow0 = mTile * 16;
    int arow = grow0 + lrow; if (arow > N - 1) arow = N - 1;
    const float* rowp = h1 + (size_t)arow * 64;

    v8f z = {0.f,0.f,0.f,0.f,0.f,0.f,0.f,0.f};
    v8f acc[2] = {z, z};

#pragma unroll
    for (int kt = 0; kt < 2; ++kt) {
        v16bf a = load_a_frag(rowp, kt * 32, half);
#pragma unroll
        for (int nt = 0; nt < 2; ++nt) {
            v16bf b = load_b_frag(sW, 2, kt, nt, lane);
            acc[nt] = __builtin_amdgcn_wmma_f32_16x16x32_bf16(
                false, a, false, b, (short)0, acc[nt], false, false);
        }
    }
#pragma unroll
    for (int nt = 0; nt < 2; ++nt)
#pragma unroll
        for (int r = 0; r < 8; ++r) {
            int grow = grow0 + half * 8 + r;
            if (grow < N) h2pre[(size_t)grow * 32 + nt * 16 + lrow] = acc[nt][r];
        }
}

// ---------------------------------------------------------------------------
// Edge scatter: hacc[dst] += hin[src] * dinv[src]*dinv[dst]
// One thread = (edge, 4 features). hin/hacc stay L2-resident (25.6 MB << 192 MB).
// ---------------------------------------------------------------------------
__global__ void gcn_agg_kernel(const int* __restrict__ src,
                               const int* __restrict__ dst,
                               const float* __restrict__ dinv,
                               const float* __restrict__ hin,
                               float* hacc, int E, int ld) {
    const int gpr = ld >> 2;                          // feature groups per row
    long long idx = (long long)blockIdx.x * blockDim.x + threadIdx.x;
    long long total = (long long)E * gpr;
    if (idx >= total) return;
    int e = (int)(idx / gpr);
    int f = (int)(idx % gpr) * 4;
    int s = src[e], d = dst[e];
    float w = dinv[s] * dinv[d];
    float4 v = *(const float4*)(hin + (size_t)s * ld + f);
    float* o = hacc + (size_t)d * ld + f;
    atomicAdd(o + 0, v.x * w);
    atomicAdd(o + 1, v.y * w);
    atomicAdd(o + 2, v.z * w);
    atomicAdd(o + 3, v.w * w);
}

// ---------------------------------------------------------------------------
// Layer-1 epilogue: add self-loop term + bias, leaky ReLU (in place on h1)
// ---------------------------------------------------------------------------
__global__ void gcn_selfbias_act(float* hacc, const float* __restrict__ hpre,
                                 const float* __restrict__ dinv,
                                 const float* __restrict__ bias, int N, int ld) {
    long long idx = (long long)blockIdx.x * blockDim.x + threadIdx.x;
    if (idx >= (long long)N * ld) return;
    int i = (int)(idx / ld);
    int f = (int)(idx % ld);
    float di = dinv[i];
    float v = hacc[idx] + hpre[idx] * di * di + bias[f];
    hacc[idx] = v > 0.f ? v : NEG_SLOPE * v;
}

// ---------------------------------------------------------------------------
// Final: self-loop + bias + leaky ReLU + log_softmax over C=32.
// One wave32 per node, lane == class; shuffle reductions for max & sum.
// ---------------------------------------------------------------------------
__global__ void gcn_final_kernel(const float* __restrict__ h2agg,
                                 const float* __restrict__ h2pre,
                                 const float* __restrict__ dinv,
                                 const float* __restrict__ b2,
                                 float* __restrict__ out, int N) {
    const int wave = threadIdx.x >> 5;
    const int lane = threadIdx.x & 31;
    const int node = blockIdx.x * 8 + wave;
    if (node >= N) return;
    float di = dinv[node];
    size_t base = (size_t)node * 32;
    float v = h2agg[base + lane] + h2pre[base + lane] * di * di + b2[lane];
    v = v > 0.f ? v : NEG_SLOPE * v;

    float m = v;
#pragma unroll
    for (int off = 16; off > 0; off >>= 1) m = fmaxf(m, __shfl_xor(m, off, 32));
    float s = __expf(v - m);
#pragma unroll
    for (int off = 16; off > 0; off >>= 1) s += __shfl_xor(s, off, 32);
    out[base + lane] = (v - m) - __logf(s);
}

// ---------------------------------------------------------------------------
// Host side
// ---------------------------------------------------------------------------
extern "C" void kernel_launch(void* const* d_in, const int* in_sizes, int n_in,
                              void* d_out, int out_size, void* d_ws, size_t ws_size,
                              hipStream_t stream) {
    const float* x  = (const float*)d_in[0];
    const int*   ei = (const int*)d_in[1];     // edge_index [2,E] int32
    const float* W1 = (const float*)d_in[2];
    const float* b1 = (const float*)d_in[3];
    const float* W2 = (const float*)d_in[4];
    const float* b2 = (const float*)d_in[5];
    float* out = (float*)d_out;

    const int N = in_sizes[0] / 256;           // F = 256
    const int E = in_sizes[1] / 2;
    const int* src = ei;
    const int* dst = ei + E;

    char* ws = (char*)d_ws;
    size_t off = 0;
    auto carve = [&](size_t bytes) {
        void* p = ws + off;
        off += (bytes + 255) & ~(size_t)255;
        return p;
    };
    float* dinv  = (float*)carve((size_t)N * 4);        // deg -> dinv
    float* hpre  = (float*)carve((size_t)N * 64 * 4);   // x @ W1
    float* h1    = (float*)carve((size_t)N * 64 * 4);   // aggregated layer-1
    float* h2pre = (float*)carve((size_t)N * 32 * 4);   // h1 @ W2
    float* h2agg = (float*)carve((size_t)N * 32 * 4);   // aggregated layer-2

    dim3 blk(256);
    const int mTiles = (N + 15) >> 4;

    gcn_init_kernel<<<(unsigned)(((long long)N * 64 + 255) / 256), blk, 0, stream>>>(
        dinv, h1, h2agg, N);
    gcn_deg_kernel<<<(E + 255) / 256, blk, 0, stream>>>(dst, dinv, E);
    gcn_rsqrt_kernel<<<(N + 255) / 256, blk, 0, stream>>>(dinv, N);

    gcn_gemm1_wmma<<<(mTiles + 7) / 8, blk, 0, stream>>>(x, W1, hpre, N);
    gcn_agg_kernel<<<(unsigned)(((long long)E * 16 + 255) / 256), blk, 0, stream>>>(
        src, dst, dinv, hpre, h1, E, 64);
    gcn_selfbias_act<<<(unsigned)(((long long)N * 64 + 255) / 256), blk, 0, stream>>>(
        h1, hpre, dinv, b1, N, 64);

    gcn_gemm2_wmma<<<(mTiles + 7) / 8, blk, 0, stream>>>(h1, W2, h2pre, N);
    gcn_agg_kernel<<<(unsigned)(((long long)E * 8 + 255) / 256), blk, 0, stream>>>(
        src, dst, dinv, h2pre, h2agg, E, 32);
    gcn_final_kernel<<<(N + 7) / 8, blk, 0, stream>>>(h2agg, h2pre, dinv, b2, out, N);
}
